// NoFoDifformer_Diff_7670811591182
// MI455X (gfx1250) — matrix operand order repaired and checked by reading, not compiled, not run on hardware
//
#include <hip/hip_runtime.h>
#include <math.h>

// ---------------------------------------------------------------------------
// CDNA5 (gfx1250) wave32 WMMA types
// ---------------------------------------------------------------------------
typedef __attribute__((ext_vector_type(16))) __bf16 bf16x16;
typedef __attribute__((ext_vector_type(8)))  float  floatx8;

#define WMMA_BF16(a, b, c) \
  __builtin_amdgcn_wmma_f32_16x16x32_bf16(false, (a), false, (b), (short)0, (c), false, false)

// Load a 16-element bf16 fragment from two 16-byte runs (A-frag: two 8-elem
// runs at k and k+16; B-frag: the same two 8-elem runs at k and k+16). Both global and
// LDS pointers work (generic address space).
__device__ __forceinline__ bf16x16 frag2(const __bf16* r0, const __bf16* r1) {
  union U { uint4 q[2]; bf16x16 v; } u;
  u.q[0] = *(const uint4*)r0;
  u.q[1] = *(const uint4*)r1;
  return u.v;
}

// ---------------------------------------------------------------------------
// Problem constants
// ---------------------------------------------------------------------------
#define NROWS 8192
#define FEATD 512
#define CDIM  128
#define NHHD  512
#define LAMBDA_INIT 0.2f

// ---------------------------------------------------------------------------
// Generic WMMA GEMM: C[M,N] = epi(A[M,K] @ B[K,N] + bias) (+res1)(+res2)
//   ACT: 0 none, 1 relu, 2 exact gelu
//   TRA: A is stored transposed (A_logical[m,k] = Aptr[k*M + m])  (for u^T @ h)
//   OUTM: 0 -> fp32 row-major, 1 -> bf16 row-major, 2 -> bf16 transposed [N,M]
// Block: 256 threads (8 waves). Tile BM=128, BN=64, BK=32.
// Each wave computes a 32x32 patch = 4 v_wmma_f32_16x16x32_bf16 per K-step.
// ---------------------------------------------------------------------------
#define BM 128
#define BN 64
#define BK 32
#define LDA 40   // padded bf16 row stride for As  (avoids LDS bank conflicts)
#define LDB 40   // padded bf16 k-stride for Bs (stored transposed [n][k])

template<int ACT, int TRA, int OUTM>
__global__ __launch_bounds__(256)
void gemm_wmma(const float* __restrict__ A, const float* __restrict__ B,
               const float* __restrict__ bias,
               const float* __restrict__ res1, const float* __restrict__ res2,
               float* __restrict__ Cf, __bf16* __restrict__ Cb,
               int M, int N, int K)
{
  __shared__ __align__(16) __bf16 As[BM * LDA];
  __shared__ __align__(16) __bf16 Bs[BN * LDB];

  const int tid  = threadIdx.x;
  const int wave = tid >> 5, lane = tid & 31;
  const int lm = lane & 15, lg = lane >> 4;
  const int bm = blockIdx.y * BM;
  const int bn = blockIdx.x * BN;
  const int wm = (wave >> 1) * 32;   // 4 waves along M
  const int wn = (wave & 1) * 32;    // 2 waves along N

  floatx8 acc[2][2] = {};

  for (int k0 = 0; k0 < K; k0 += BK) {
    // --- stage A tile (BM x BK) as bf16, converting on the fly ---
#pragma unroll
    for (int i = 0; i < (BM * BK) / 256; ++i) {
      int idx = tid + i * 256;
      if (TRA) {
        int c = idx >> 7, r = idx & 127;                  // coalesce along m
        As[r * LDA + c] = (__bf16)A[(size_t)(k0 + c) * M + bm + r];
      } else {
        int r = idx >> 5, c = idx & 31;                   // coalesce along k
        As[r * LDA + c] = (__bf16)A[(size_t)(bm + r) * K + k0 + c];
      }
    }
    // --- stage B tile (BK x BN) transposed to [n][k] ---
#pragma unroll
    for (int i = 0; i < (BN * BK) / 256; ++i) {
      int idx = tid + i * 256;
      int kk = idx >> 6, nn = idx & 63;                   // coalesce along n
      Bs[nn * LDB + kk] = (__bf16)B[(size_t)(k0 + kk) * N + bn + nn];
    }
    __syncthreads();

#pragma unroll
    for (int i = 0; i < 2; ++i) {
      const __bf16* ap = &As[(wm + i * 16 + lm) * LDA];
      bf16x16 af = frag2(ap + lg * 8, ap + 16 + lg * 8);  // A 16x32 layout
#pragma unroll
      for (int j = 0; j < 2; ++j) {
        const __bf16* bp = &Bs[(wn + j * 16 + lm) * LDB];
        bf16x16 bfm = frag2(bp + lg * 8, bp + 16 + lg * 8);  // B 32x16 layout
        acc[i][j] = WMMA_BF16(af, bfm, acc[i][j]);
      }
    }
    __syncthreads();
  }

  // --- epilogue: bias + activation + residuals, store ---
#pragma unroll
  for (int i = 0; i < 2; ++i)
#pragma unroll
    for (int j = 0; j < 2; ++j)
#pragma unroll
      for (int v = 0; v < 8; ++v) {
        int gm = bm + wm + i * 16 + lg * 8 + v;           // C/D: m=(l>>4)*8+j
        int gn = bn + wn + j * 16 + lm;                   //      n=l&15
        float x = acc[i][j][v];
        if (bias) x += bias[gn];
        if (ACT == 1) x = fmaxf(x, 0.0f);
        else if (ACT == 2) x = 0.5f * x * (1.0f + erff(x * 0.70710678118f));
        if (res1) x += res1[(size_t)gm * N + gn];
        if (res2) x += res2[(size_t)gm * N + gn];
        if (OUTM == 0)      Cf[(size_t)gm * N + gn] = x;
        else if (OUTM == 1) Cb[(size_t)gm * N + gn] = (__bf16)x;
        else                Cb[(size_t)gn * M + gm] = (__bf16)x;  // transposed
      }
}

// ---------------------------------------------------------------------------
// Differential flash attention:
//   out = softmax(q1@k1^T)@v - lam * softmax(q2@k2^T)@v      (no 1/sqrt scale)
// Block = 256 threads = 8 waves, handles 16 query rows.
//   Phase 1: waves partition D=512 into 64-wide slices, each computes a
//            partial S[16,32] per branch with WMMA, reduced in LDS (ds_add_f32)
//   Phase 2: threads 0..31 run the online-softmax update (one (branch,row) each)
//   Phase 3: waves partition 512 output columns (64 each), rescale acc,
//            P@V via WMMA against transposed bf16 V.
// ---------------------------------------------------------------------------
__global__ __launch_bounds__(256)
void diff_attn(const __bf16* __restrict__ q1, const __bf16* __restrict__ k1,
               const __bf16* __restrict__ q2, const __bf16* __restrict__ k2,
               const __bf16* __restrict__ vT, const float* __restrict__ lam_p,
               float* __restrict__ out, int N)
{
  __shared__ __align__(16) float  S[2][16][32];
  __shared__ __align__(16) __bf16 P[2][16][40];
  __shared__ float alpha_s[2][16];
  __shared__ float mrun[2][16], lrun[2][16];

  const int tid = threadIdx.x, wave = tid >> 5, lane = tid & 31;
  const int lm = lane & 15, lg = lane >> 4;
  const int q0    = blockIdx.x * 16;
  const int dbase = wave * 64;   // this wave's slice of the 512-d dot product
  const int cbase = wave * 64;   // this wave's slice of the 512 output columns

  // preload Q fragments for this wave's d-slice (2 k-steps x 2 branches)
  bf16x16 qf[2][2];
#pragma unroll
  for (int b = 0; b < 2; ++b) {
    const __bf16* Q = b ? q2 : q1;
#pragma unroll
    for (int s = 0; s < 2; ++s) {
      const __bf16* p0 = Q + (size_t)(q0 + lm) * NHHD + dbase + s * 32 + lg * 8;
      qf[b][s] = frag2(p0, p0 + 16);
    }
  }
  if (tid < 32) {
    int b = tid >> 4, r = tid & 15;
    mrun[b][r] = -1e30f;
    lrun[b][r] = 0.0f;
  }

  floatx8 acc[2][4] = {};   // [branch][16-col tile] over 16 rows x 64 cols

  for (int kb = 0; kb < N; kb += 32) {
    // zero logit scratch (1024 floats by 256 threads)
#pragma unroll
    for (int i = 0; i < 4; ++i) ((float*)S)[tid + i * 256] = 0.0f;
    __syncthreads();

    // partial logits for this wave's d-slice -> LDS float atomics
#pragma unroll
    for (int b = 0; b < 2; ++b) {
      const __bf16* Kmat = b ? k2 : k1;
#pragma unroll
      for (int nt = 0; nt < 2; ++nt) {
        floatx8 sacc = {};
#pragma unroll
        for (int s = 0; s < 2; ++s) {
          // B[d,n] = K[kb+n][d] : two 8-elem runs at d and d+16 per lane
          const __bf16* kp = Kmat + (size_t)(kb + nt * 16 + lm) * NHHD
                                  + dbase + s * 32 + lg * 8;
          bf16x16 kf = frag2(kp, kp + 16);
          sacc = WMMA_BF16(qf[b][s], kf, sacc);
        }
#pragma unroll
        for (int v = 0; v < 8; ++v)
          atomicAdd(&S[b][lg * 8 + v][nt * 16 + lm], sacc[v]);
      }
    }
    __syncthreads();

    // online softmax bookkeeping: one (branch,row) per thread
    if (tid < 32) {
      int b = tid >> 4, r = tid & 15;
      float mold = mrun[b][r];
      float mt = mold;
#pragma unroll
      for (int c = 0; c < 32; ++c) mt = fmaxf(mt, S[b][r][c]);
      float al = __expf(mold - mt);
      float ps = 0.0f;
#pragma unroll
      for (int c = 0; c < 32; ++c) {
        float pv = __expf(S[b][r][c] - mt);
        P[b][r][c] = (__bf16)pv;
        ps += pv;
      }
      mrun[b][r]    = mt;
      lrun[b][r]    = lrun[b][r] * al + ps;
      alpha_s[b][r] = al;
    }
    __syncthreads();

    // rescale accumulators and add P @ V for this wave's 64 columns
#pragma unroll
    for (int b = 0; b < 2; ++b) {
      const __bf16* pp = &P[b][lm][0];
      bf16x16 pf = frag2(pp + lg * 8, pp + 16 + lg * 8);  // A 16x32 layout
#pragma unroll
      for (int nt = 0; nt < 4; ++nt) {
#pragma unroll
        for (int v = 0; v < 8; ++v)
          acc[b][nt][v] *= alpha_s[b][lg * 8 + v];
        // B[k,n] = V[kb+k][col] = vT[col][kb+k] : two 8-elem runs at k and k+16 per lane
        const __bf16* vp = vT + (size_t)(cbase + nt * 16 + lm) * N + kb + lg * 8;
        bf16x16 vf = frag2(vp, vp + 16);
        acc[b][nt] = WMMA_BF16(pf, vf, acc[b][nt]);
      }
    }
    __syncthreads();
  }

  // epilogue: out = acc1/l1 - lam * acc2/l2
  const float lam = *lam_p;
#pragma unroll
  for (int nt = 0; nt < 4; ++nt)
#pragma unroll
    for (int v = 0; v < 8; ++v) {
      int r = lg * 8 + v;
      int c = cbase + nt * 16 + lm;
      float o = acc[0][nt][v] / lrun[0][r] - lam * (acc[1][nt][v] / lrun[1][r]);
      out[(size_t)(q0 + r) * NHHD + c] = o;
    }
}

// ---------------------------------------------------------------------------
// LayerNorm: one wave per row.  Y = ((X-mu)/sqrt(var+eps) * g + b) * post
// ---------------------------------------------------------------------------
__global__ __launch_bounds__(256)
void ln_rows(const float* __restrict__ X, const float* __restrict__ g,
             const float* __restrict__ b, float* __restrict__ Y,
             int W, float post)
{
  int wave = threadIdx.x >> 5, lane = threadIdx.x & 31;
  int row = blockIdx.x * 8 + wave;
  const float* x = X + (size_t)row * W;
  float s = 0.0f, s2 = 0.0f;
  for (int c = lane; c < W; c += 32) { float v = x[c]; s += v; s2 += v * v; }
#pragma unroll
  for (int m = 16; m; m >>= 1) {
    s  += __shfl_xor(s,  m, 32);
    s2 += __shfl_xor(s2, m, 32);
  }
  float mean = s / (float)W;
  float var  = s2 / (float)W - mean * mean;
  float inv  = rsqrtf(var + 1e-5f);
  float* y = Y + (size_t)row * W;
  for (int c = lane; c < W; c += 32)
    y[c] = ((x[c] - mean) * inv * g[c] + b[c]) * post;
}

// ---------------------------------------------------------------------------
// SineEncoding -> new_e[N] : K=10 powers, half=16, eig_w[10][33], alpha_w[10]
// ---------------------------------------------------------------------------
__global__ void sine_newe(const float* __restrict__ e, const float* __restrict__ eig_w,
                          const float* __restrict__ eig_b, const float* __restrict__ alpha_w,
                          float* __restrict__ new_e, int N)
{
  int n = blockIdx.x * blockDim.x + threadIdx.x;
  if (n >= N) return;
  float ev = e[n];
  float outv = 0.0f, p = 1.0f;
#pragma unroll
  for (int k = 0; k < 10; ++k) {
    p *= ev;                           // e^(k+1)
    const float* w = eig_w + k * 33;
    float s = w[0] + eig_b[k];         // feat[...,0] = 1
#pragma unroll
    for (int i = 1; i <= 16; ++i) {
      float ang = p * (float)i;
      float sv, cv;
      __sincosf(ang, &sv, &cv);
      s += w[i] * sv + w[16 + i] * cv;
    }
    outv += s * alpha_w[k];
  }
  new_e[n] = outv;
}

// lam_full = exp(sum lq1*lk1) - exp(sum lq2*lk2) + LAMBDA_INIT
__global__ void lambda_kernel(const float* lq1, const float* lk1,
                              const float* lq2, const float* lk2, float* lam)
{
  int lane = threadIdx.x;
  float s1 = 0.0f, s2 = 0.0f;
  for (int c = lane; c < CDIM; c += 32) { s1 += lq1[c] * lk1[c]; s2 += lq2[c] * lk2[c]; }
#pragma unroll
  for (int m = 16; m; m >>= 1) {
    s1 += __shfl_xor(s1, m, 32);
    s2 += __shfl_xor(s2, m, 32);
  }
  if (lane == 0) *lam = expf(s1) - expf(s2) + LAMBDA_INIT;
}

// Y[r,c] = X[r,c] * s[r]   (row scaling by new_e before u @ (...))
__global__ void rowscale_kernel(const float* __restrict__ X, const float* __restrict__ s,
                                float* __restrict__ Y, int total, int C)
{
  int i = blockIdx.x * blockDim.x + threadIdx.x;
  if (i < total) Y[i] = X[i] * s[i / C];
}

// ---------------------------------------------------------------------------
// Host orchestration
// ---------------------------------------------------------------------------
template<int ACT, int TRA, int OUTM>
static void launch_gemm(const float* A, const float* B, const float* bias,
                        const float* r1, const float* r2,
                        float* Cf, __bf16* Cb, int M, int N, int K, hipStream_t s)
{
  dim3 grid(N / BN, M / BM), block(256);
  gemm_wmma<ACT, TRA, OUTM><<<grid, block, 0, s>>>(A, B, bias, r1, r2, Cf, Cb, M, N, K);
}

extern "C" void kernel_launch(void* const* d_in, const int* in_sizes, int n_in,
                              void* d_out, int out_size, void* d_ws, size_t ws_size,
                              hipStream_t stream)
{
  const float* e      = (const float*)d_in[0];
  const float* u      = (const float*)d_in[1];
  const float* x      = (const float*)d_in[2];
  const float* fe_w1  = (const float*)d_in[3];
  const float* fe_b1  = (const float*)d_in[4];
  const float* fe_w2  = (const float*)d_in[5];
  const float* fe_b2  = (const float*)d_in[6];
  const float* eig_w  = (const float*)d_in[7];
  const float* eig_b  = (const float*)d_in[8];
  const float* alpha_w= (const float*)d_in[9];
  const float* mha_g  = (const float*)d_in[10];
  const float* mha_b  = (const float*)d_in[11];
  const float* wq1    = (const float*)d_in[12];
  const float* bq1    = (const float*)d_in[13];
  const float* wk1    = (const float*)d_in[14];
  const float* bk1    = (const float*)d_in[15];
  const float* wq2    = (const float*)d_in[16];
  const float* bq2    = (const float*)d_in[17];
  const float* wk2    = (const float*)d_in[18];
  const float* bk2    = (const float*)d_in[19];
  const float* wv     = (const float*)d_in[20];
  const float* bv     = (const float*)d_in[21];
  const float* ln_g   = (const float*)d_in[22];
  const float* ln_b   = (const float*)d_in[23];
  const float* wo     = (const float*)d_in[24];
  const float* bo     = (const float*)d_in[25];
  const float* lq1    = (const float*)d_in[26];
  const float* lk1    = (const float*)d_in[27];
  const float* lq2    = (const float*)d_in[28];
  const float* lk2    = (const float*)d_in[29];
  const float* ffn_g  = (const float*)d_in[30];
  const float* ffn_b  = (const float*)d_in[31];
  const float* fw1    = (const float*)d_in[32];
  const float* fb1    = (const float*)d_in[33];
  const float* fw2    = (const float*)d_in[34];
  const float* fb2    = (const float*)d_in[35];

  // workspace carve-up (256-byte aligned slabs)
  char* ws = (char*)d_ws;
  size_t off = 0;
  auto carve = [&](size_t bytes) -> void* {
    void* p = ws + off;
    off += (bytes + 255) & ~(size_t)255;
    return p;
  };
  const size_t fNC   = (size_t)NROWS * CDIM  * sizeof(float);   // 4 MB
  const size_t fNH   = (size_t)NROWS * NHHD  * sizeof(float);   // 16 MB
  const size_t bNH   = (size_t)NROWS * NHHD  * sizeof(__bf16);  // 8 MB

  float*  new_e   = (float*)carve((size_t)NROWS * sizeof(float));
  float*  lam     = (float*)carve(256);
  float*  T1      = (float*)carve(fNC);
  float*  h       = (float*)carve(fNC);
  float*  utx     = (float*)carve(fNC);
  float*  b2s     = (float*)carve(fNC);
  float*  hfur    = (float*)carve(fNC);
  float*  mhah    = (float*)carve(fNC);
  __bf16* q1b     = (__bf16*)carve(bNH);
  __bf16* k1b     = (__bf16*)carve(bNH);
  __bf16* q2b     = (__bf16*)carve(bNH);
  __bf16* k2b     = (__bf16*)carve(bNH);
  __bf16* vbT     = (__bf16*)carve(bNH);
  float*  attn_r  = (float*)carve(fNH);
  float*  attn_ln = (float*)carve(fNH);
  float*  mhah2   = (float*)carve(fNC);
  float*  ffnh    = (float*)carve(fNC);
  float*  T2      = (float*)carve(fNC);
  float*  out     = (float*)d_out;

  // 1) SineEncoding -> new_e ; lambda scalar
  sine_newe<<<NROWS / 256, 256, 0, stream>>>(e, eig_w, eig_b, alpha_w, new_e, NROWS);
  lambda_kernel<<<1, 32, 0, stream>>>(lq1, lk1, lq2, lk2, lam);

  // 2) feat encoder: T1 = relu(x @ fe_w1 + b1) ; h = T1 @ fe_w2 + b2
  launch_gemm<1, 0, 0>(x,  fe_w1, fe_b1, nullptr, nullptr, T1, nullptr, NROWS, CDIM, FEATD, stream);
  launch_gemm<0, 0, 0>(T1, fe_w2, fe_b2, nullptr, nullptr, h,  nullptr, NROWS, CDIM, CDIM,  stream);

  // 3) spectral: utx = u^T @ h ; h_fur = u @ (new_e * utx)
  launch_gemm<0, 1, 0>(u, h,   nullptr, nullptr, nullptr, utx,  nullptr, NROWS, CDIM, NROWS, stream);
  rowscale_kernel<<<(NROWS * CDIM) / 256, 256, 0, stream>>>(utx, new_e, b2s, NROWS * CDIM, CDIM);
  launch_gemm<0, 0, 0>(u, b2s, nullptr, nullptr, nullptr, hfur, nullptr, NROWS, CDIM, NROWS, stream);

  // 4) mha_h = LN(h)
  ln_rows<<<NROWS / 8, 256, 0, stream>>>(h, mha_g, mha_b, mhah, CDIM, 1.0f);

  // 5) projections -> bf16 (V transposed for attention fragments)
  launch_gemm<0, 0, 1>(mhah, wq1, bq1, nullptr, nullptr, nullptr, q1b, NROWS, NHHD, CDIM, stream);
  launch_gemm<0, 0, 1>(mhah, wk1, bk1, nullptr, nullptr, nullptr, k1b, NROWS, NHHD, CDIM, stream);
  launch_gemm<0, 0, 1>(mhah, wq2, bq2, nullptr, nullptr, nullptr, q2b, NROWS, NHHD, CDIM, stream);
  launch_gemm<0, 0, 1>(mhah, wk2, bk2, nullptr, nullptr, nullptr, k2b, NROWS, NHHD, CDIM, stream);
  launch_gemm<0, 0, 2>(mhah, wv,  bv,  nullptr, nullptr, nullptr, vbT, NROWS, NHHD, CDIM, stream);

  // 6) differential flash attention
  diff_attn<<<NROWS / 16, 256, 0, stream>>>(q1b, k1b, q2b, k2b, vbT, lam, attn_r, NROWS);

  // 7) attn = LN(attn_raw) * (1 - lambda_init)
  ln_rows<<<NROWS / 8, 256, 0, stream>>>(attn_r, ln_g, ln_b, attn_ln, NHHD, 1.0f - LAMBDA_INIT);

  // 8) mha_h_ = attn @ wo + bo + h + h_fur
  launch_gemm<0, 0, 0>(attn_ln, wo, bo, h, hfur, mhah2, nullptr, NROWS, CDIM, NHHD, stream);

  // 9) FFN: ffn_h = LN(mha_h_) ; out = gelu(ffn_h@fw1+fb1)@fw2 + fb2 + mha_h_
  ln_rows<<<NROWS / 8, 256, 0, stream>>>(mhah2, ffn_g, ffn_b, ffnh, CDIM, 1.0f);
  launch_gemm<2, 0, 0>(ffnh, fw1, fb1, nullptr, nullptr, T2, nullptr, NROWS, CDIM, CDIM, stream);
  launch_gemm<0, 0, 0>(T2,   fw2, fb2, mhah2,   nullptr, out, nullptr, NROWS, CDIM, CDIM, stream);
}
